// MambaRecurrence_55448027791587
// MI455X (gfx1250) — matrix operand order repaired and checked by reading, not compiled, not run on hardware
//
#include <hip/hip_runtime.h>
#include <hip/hip_bf16.h>
#include <cstdint>
#include <math.h>

// ---------------- problem dims (match reference) ----------------
#define D_MODEL 2048
#define D_STATE 16
#define DT_RANK 128
#define NX      (DT_RANK + 2 * D_STATE)   // 160
#define BATCH   4
#define LSEQ    2048
#define MROWS   (BATCH * LSEQ)            // 8192

#define NT1 (NX / 16)        // 10 N-tiles for GEMM1
#define KI1 (D_MODEL / 32)   // 64 K-steps for GEMM1
#define NT2 (D_MODEL / 16)   // 128 N-tiles for GEMM2
#define KI2 (DT_RANK / 32)   // 4 K-steps for GEMM2
#define G1FRAG (KI1 * NT1 * 32 * 8)   // dwords of packed B frags, GEMM1
#define G2FRAG (KI2 * NT2 * 32 * 8)   // dwords of packed B frags, GEMM2

typedef __attribute__((ext_vector_type(16))) __bf16 v16bf;
typedef __attribute__((ext_vector_type(8)))  float  v8f;

union V16BF { v16bf v; uint4 q[2]; };

// ---------------- helpers ----------------
static __device__ inline uint32_t bfpair(__bf16 a, __bf16 b) {
    uint16_t ua, ub;
    __builtin_memcpy(&ua, &a, 2);
    __builtin_memcpy(&ub, &b, 2);
    return (uint32_t)ua | ((uint32_t)ub << 16);
}

static __device__ inline float softplusf(float x) {
    // numerically stable: max(x,0) + log1p(exp(-|x|))
    return fmaxf(x, 0.0f) + log1pf(__expf(-fabsf(x)));
}

// Build A-fragment (16x32 bf16, contiguous k = 16*hlf + s per lane) from an fp32 row,
// as hi + lo split-bf16 for near-fp32 accuracy.
static __device__ inline void load_a_frag(const float* __restrict__ row, int kbase,
                                          int hlf, v16bf& ahi, v16bf& alo) {
#pragma unroll
    for (int j = 0; j < 8; ++j) {
        int kk = kbase + hlf * 16 + 2 * j;
        float f0 = row[kk], f1 = row[kk + 1];
        __bf16 h0 = (__bf16)f0, h1 = (__bf16)f1;
        ahi[2 * j]     = h0;
        ahi[2 * j + 1] = h1;
        alo[2 * j]     = (__bf16)(f0 - (float)h0);
        alo[2 * j + 1] = (__bf16)(f1 - (float)h1);
    }
}

// 3-term split-bf16 accumulate: C += Ahi*Bhi + Ahi*Blo + Alo*Bhi
static __device__ inline v8f wmma3(v16bf ahi, v16bf alo,
                                   const uint32_t* __restrict__ pbhi,
                                   const uint32_t* __restrict__ pblo,
                                   int fragbase, v8f c) {
    V16BF bh, bl;
    const uint4* ph = (const uint4*)(pbhi + fragbase);
    const uint4* pl = (const uint4*)(pblo + fragbase);
    bh.q[0] = ph[0]; bh.q[1] = ph[1];
    bl.q[0] = pl[0]; bl.q[1] = pl[1];
    c = __builtin_amdgcn_wmma_f32_16x16x32_bf16(false, ahi, false, bh.v, (short)0, c, false, false);
    c = __builtin_amdgcn_wmma_f32_16x16x32_bf16(false, alo, false, bh.v, (short)0, c, false, false);
    c = __builtin_amdgcn_wmma_f32_16x16x32_bf16(false, ahi, false, bl.v, (short)0, c, false, false);
    return c;
}

// ---------------- kernel 0: pack weights into B-fragment layout ----------------
// B tile (32x16): lane = hlf*16 + n(0..15); dword j holds K = hlf*16 + 2j, 2j+1.
// B[k][n] = W[n][k] (we compute X @ W^T). Stored as hi/lo split bf16 pairs.
__global__ __launch_bounds__(256)
void pack_weights(const float* __restrict__ W1, const float* __restrict__ W2,
                  uint32_t* __restrict__ b1hi, uint32_t* __restrict__ b1lo,
                  uint32_t* __restrict__ b2hi, uint32_t* __restrict__ b2lo) {
    int p = blockIdx.x * 256 + threadIdx.x;
    if (p >= G1FRAG + G2FRAG) return;
    const float* W; uint32_t *oh, *ol; int nt, ki, ld, q;
    if (p < G1FRAG) {
        q = p; W = W1; oh = b1hi; ol = b1lo; ld = D_MODEL;
        int rem = q >> 8;             // / (32 lanes * 8 dwords)
        nt = rem % NT1; ki = rem / NT1;
    } else {
        q = p - G1FRAG; W = W2; oh = b2hi; ol = b2lo; ld = DT_RANK;
        int rem = q >> 8;
        nt = rem % NT2; ki = rem / NT2;
    }
    int j    = q & 7;
    int lane = (q >> 3) & 31;
    int n    = nt * 16 + (lane & 15);
    int hlf  = lane >> 4;
    int kk   = ki * 32 + hlf * 16 + 2 * j;
    float f0 = W[n * ld + kk], f1 = W[n * ld + kk + 1];
    __bf16 h0 = (__bf16)f0, h1 = (__bf16)f1;
    oh[q] = bfpair(h0, h1);
    ol[q] = bfpair((__bf16)(f0 - (float)h0), (__bf16)(f1 - (float)h1));
}

// ---------------- kernel 1: x_dbl = H @ x_proj_w^T  (8192x2048 * 2048x160) ----
// One wave per 16-row M-tile, all 10 N-tiles kept in 80 accumulator VGPRs:
// H streamed from HBM exactly once; W fragments are L2-resident vector loads.
__global__ __launch_bounds__(256)
void gemm1_xproj(const float* __restrict__ X,
                 const uint32_t* __restrict__ b1hi, const uint32_t* __restrict__ b1lo,
                 float* __restrict__ xdbl) {
    int wave = threadIdx.x >> 5;
    int lane = threadIdx.x & 31;
    int mt   = blockIdx.x * 8 + wave;          // 0..511
    int hlf  = lane >> 4;
    const float* arow = X + (size_t)(mt * 16 + (lane & 15)) * D_MODEL;

    v8f zero = {0.f, 0.f, 0.f, 0.f, 0.f, 0.f, 0.f, 0.f};
    v8f acc[NT1];
#pragma unroll
    for (int i = 0; i < NT1; ++i) acc[i] = zero;

    for (int ki = 0; ki < KI1; ++ki) {
        v16bf ahi, alo;
        load_a_frag(arow, ki * 32, hlf, ahi, alo);
#pragma unroll
        for (int nt = 0; nt < NT1; ++nt) {
            int fb = ((ki * NT1 + nt) * 32 + lane) * 8;
            acc[nt] = wmma3(ahi, alo, b1hi, b1lo, fb, acc[nt]);
        }
    }
    // D layout: VGPR r, lane = hlf*16 + n  ->  element (M = r + 8*hlf, N = n)
#pragma unroll
    for (int nt = 0; nt < NT1; ++nt) {
        int n = nt * 16 + (lane & 15);
#pragma unroll
        for (int r = 0; r < 8; ++r) {
            int row = mt * 16 + r + 8 * hlf;
            xdbl[row * NX + n] = acc[nt][r];
        }
    }
}

// ---------------- kernel 2: delta = softplus(dt_r @ dt_proj_w^T + b) ----------
// Writes delta into d_out (same shape as y); the scan reads it and overwrites
// in place. Each wave computes a 16x64 slab (4 accumulators), K = 128.
__global__ __launch_bounds__(256)
void gemm2_delta(const float* __restrict__ xdbl,
                 const uint32_t* __restrict__ b2hi, const uint32_t* __restrict__ b2lo,
                 const float* __restrict__ bias, float* __restrict__ delta) {
    int wave = threadIdx.x >> 5;
    int lane = threadIdx.x & 31;
    int mt   = blockIdx.x;                      // 0..511
    int slab = blockIdx.y * 8 + wave;           // 0..31 (64 cols each)
    int hlf  = lane >> 4;
    const float* arow = xdbl + (size_t)(mt * 16 + (lane & 15)) * NX;  // dt_r cols 0..127

    v8f zero = {0.f, 0.f, 0.f, 0.f, 0.f, 0.f, 0.f, 0.f};
    v8f acc[4];
#pragma unroll
    for (int i = 0; i < 4; ++i) acc[i] = zero;

#pragma unroll
    for (int ki = 0; ki < KI2; ++ki) {
        v16bf ahi, alo;
        load_a_frag(arow, ki * 32, hlf, ahi, alo);
#pragma unroll
        for (int nt2 = 0; nt2 < 4; ++nt2) {
            int nt = slab * 4 + nt2;
            int fb = ((ki * NT2 + nt) * 32 + lane) * 8;
            acc[nt2] = wmma3(ahi, alo, b2hi, b2lo, fb, acc[nt2]);
        }
    }
#pragma unroll
    for (int nt2 = 0; nt2 < 4; ++nt2) {
        int n = slab * 64 + nt2 * 16 + (lane & 15);
        float bv = bias[n];
#pragma unroll
        for (int r = 0; r < 8; ++r) {
            int row = mt * 16 + r + 8 * hlf;
            delta[(size_t)row * D_MODEL + n] = softplusf(acc[nt2][r] + bv);
        }
    }
}

// ---------------- kernel 3: selective scan -----------------------------------
// One lane per (b,d) channel. 16 h-states + 16 precomputed A*log2e in VGPRs.
// delta is read from Y and y is written back to the same slot after use.
// Next-step u/delta/B/C loads are software double-buffered to hide latency.
__global__ __launch_bounds__(64)
void scan_kernel(const float* __restrict__ U, const float* __restrict__ xdbl,
                 const float* __restrict__ A_log, const float* __restrict__ Dv,
                 float* __restrict__ Y) {
    const int d = blockIdx.x * 64 + threadIdx.x;
    const int b = blockIdx.y;

    float a2[D_STATE], h[D_STATE];
#pragma unroll
    for (int n = 0; n < D_STATE; ++n) {
        // A = -exp(A_log); pre-scale by log2(e) so the step uses hw exp2
        a2[n] = -__expf(A_log[d * D_STATE + n]) * 1.4426950408889634f;
        h[n]  = 0.f;
    }
    const float dcoef = Dv[d];

    const float* __restrict__ up = U + ((size_t)b * LSEQ) * D_MODEL + d;
    float* __restrict__ yp       = Y + ((size_t)b * LSEQ) * D_MODEL + d;
    const float4* __restrict__ bc =
        (const float4*)(xdbl + ((size_t)b * LSEQ) * NX + DT_RANK); // B(16)|C(16)

    // prologue: load t = 0
    float u_c  = up[0];
    float dl_c = yp[0];
    float4 bv0 = bc[0], bv1 = bc[1], bv2 = bc[2], bv3 = bc[3];
    float4 cv0 = bc[4], cv1 = bc[5], cv2 = bc[6], cv3 = bc[7];

    for (int t = 0; t < LSEQ; ++t) {
        float u = u_c, dl = dl_c;
        float4 b0 = bv0, b1 = bv1, b2 = bv2, b3 = bv3;
        float4 c0 = cv0, c1 = cv1, c2 = cv2, c3 = cv3;
        if (t + 1 < LSEQ) {   // prefetch next step
            u_c  = up[(size_t)(t + 1) * D_MODEL];
            dl_c = yp[(size_t)(t + 1) * D_MODEL];
            const float4* nb = bc + (size_t)(t + 1) * (NX / 4);
            bv0 = nb[0]; bv1 = nb[1]; bv2 = nb[2]; bv3 = nb[3];
            cv0 = nb[4]; cv1 = nb[5]; cv2 = nb[6]; cv3 = nb[7];
        }
        float du = dl * u;
        float y  = 0.f;
#define SSTEP(ni, Bc, Cc)                                        \
        { float e = __builtin_amdgcn_exp2f(dl * a2[ni]);         \
          h[ni] = fmaf(e, h[ni], du * (Bc));                     \
          y = fmaf(h[ni], (Cc), y); }
        SSTEP(0,  b0.x, c0.x) SSTEP(1,  b0.y, c0.y) SSTEP(2,  b0.z, c0.z) SSTEP(3,  b0.w, c0.w)
        SSTEP(4,  b1.x, c1.x) SSTEP(5,  b1.y, c1.y) SSTEP(6,  b1.z, c1.z) SSTEP(7,  b1.w, c1.w)
        SSTEP(8,  b2.x, c2.x) SSTEP(9,  b2.y, c2.y) SSTEP(10, b2.z, c2.z) SSTEP(11, b2.w, c2.w)
        SSTEP(12, b3.x, c3.x) SSTEP(13, b3.y, c3.y) SSTEP(14, b3.z, c3.z) SSTEP(15, b3.w, c3.w)
#undef SSTEP
        yp[(size_t)t * D_MODEL] = fmaf(u, dcoef, y);   // overwrite delta slot
    }
}

// ---------------- host launcher ----------------------------------------------
extern "C" void kernel_launch(void* const* d_in, const int* in_sizes, int n_in,
                              void* d_out, int out_size, void* d_ws, size_t ws_size,
                              hipStream_t stream) {
    (void)in_sizes; (void)n_in; (void)out_size; (void)ws_size;
    const float* X     = (const float*)d_in[0];   // hidden_states (4,2048,2048)
    const float* W1    = (const float*)d_in[1];   // x_proj_w (160,2048)
    const float* W2    = (const float*)d_in[2];   // dt_proj_w (2048,128)
    const float* bias  = (const float*)d_in[3];   // dt_proj_b (2048,)
    const float* A_log = (const float*)d_in[4];   // (2048,16)
    const float* Dv    = (const float*)d_in[5];   // (2048,)
    float* out = (float*)d_out;

    char* ws = (char*)d_ws;
    float*    xdbl = (float*)ws;                                  // 8192*160*4 = 5,242,880 B
    uint32_t* b1hi = (uint32_t*)(ws + (size_t)MROWS * NX * 4);
    uint32_t* b1lo = b1hi + G1FRAG;
    uint32_t* b2hi = b1lo + G1FRAG;
    uint32_t* b2lo = b2hi + G2FRAG;                               // total ws ~7.6 MB

    const int totalPack = G1FRAG + G2FRAG;                        // 294912
    pack_weights<<<(totalPack + 255) / 256, 256, 0, stream>>>(W1, W2, b1hi, b1lo, b2hi, b2lo);
    gemm1_xproj<<<MROWS / 16 / 8, 256, 0, stream>>>(X, b1hi, b1lo, xdbl);
    gemm2_delta<<<dim3(MROWS / 16, 4), 256, 0, stream>>>(xdbl, b2hi, b2lo, bias, out);
    scan_kernel<<<dim3(D_MODEL / 64, BATCH), 64, 0, stream>>>(X, xdbl, A_log, Dv, out);
}
